// MultiHeadAttention_34548716929853
// MI455X (gfx1250) — matrix operand order repaired and checked by reading, not compiled, not run on hardware
//
#include <hip/hip_runtime.h>

// ---------------- types ----------------
typedef __attribute__((ext_vector_type(16))) _Float16 v16h;
typedef __attribute__((ext_vector_type(8)))  _Float16 v8h;
typedef __attribute__((ext_vector_type(8)))  float    v8f;
typedef __attribute__((vector_size(16)))     int      v4i;   // for async-LDS builtin

#define DEV __device__ __forceinline__

static constexpr int BATCH   = 2;
static constexpr int SEQ     = 2048;
static constexpr int DMODEL  = 2048;
static constexpr int NHEADS  = 16;
static constexpr int DK      = 128;
static constexpr int MROWS   = BATCH * SEQ;      // 4096

// ---- CDNA5 async global->LDS copy (guarded; falls back to sync copy) ----
#if defined(__has_builtin)
#if __has_builtin(__builtin_amdgcn_global_load_async_to_lds_b128)
#define HAVE_ASYNC_LDS 1
#endif
#endif
#ifndef HAVE_ASYNC_LDS
#define HAVE_ASYNC_LDS 0
#endif

#define GLOBAL_AS __attribute__((address_space(1)))
#define LDS_AS    __attribute__((address_space(3)))

DEV void stage_b128(const _Float16* g, _Float16* l) {
#if HAVE_ASYNC_LDS
    __builtin_amdgcn_global_load_async_to_lds_b128(
        (GLOBAL_AS v4i*)const_cast<_Float16*>(g),
        (LDS_AS v4i*)l, 0, 0);
#else
    *(v8h*)l = *(const v8h*)g;
#endif
}

DEV void stage_fence() {     // make staged LDS data visible to all waves
#if HAVE_ASYNC_LDS
    asm volatile("s_wait_asynccnt 0x0" ::: "memory");
#endif
    __syncthreads();
}

DEV v16h combine16(v8h lo, v8h hi) {
    v16h a;
#pragma unroll
    for (int e = 0; e < 8; ++e) { a[e] = lo[e]; a[8 + e] = hi[e]; }
    return a;
}

DEV v8f wmma_f16(v16h a, v16h b, v8f c) {
    // D = A*B + C, fp32 accumulate (v_wmma_f32_16x16x32_f16)
    return __builtin_amdgcn_wmma_f32_16x16x32_f16(
        /*neg_a=*/false, a, /*neg_b=*/false, b,
        /*c_mod=*/(short)0, c, /*reuse_a=*/false, /*reuse_b=*/false);
}

DEV v8h cvt_f32x8_to_h8(float4 f0, float4 f1) {
    v8h h;
    h[0] = (_Float16)f0.x; h[1] = (_Float16)f0.y;
    h[2] = (_Float16)f0.z; h[3] = (_Float16)f0.w;
    h[4] = (_Float16)f1.x; h[5] = (_Float16)f1.y;
    h[6] = (_Float16)f1.z; h[7] = (_Float16)f1.w;
    return h;
}

// =====================================================================
// One-shot fp32 -> f16 conversion (x and the four weight matrices).
// 8 elements / thread, vectorized b128 load + b128 store.
// =====================================================================
__global__ __launch_bounds__(256)
void f32_to_f16_kernel(const float* __restrict__ src, _Float16* __restrict__ dst) {
    const size_t i = ((size_t)blockIdx.x * blockDim.x + threadIdx.x) * 8;
    const float4* s = (const float4*)(src + i);
    float4 f0 = s[0], f1 = s[1];
    *(v8h*)(dst + i) = cvt_f32x8_to_h8(f0, f1);
}

// =====================================================================
// GEMM:  C[M,N] = A[M,K] * W[N,K]^T + bias   (M=4096, N=K=2048, all f16)
// Block tile 128x128, K-step 32, 256 threads = 8 waves (2 x 4 wave grid),
// each wave computes 64x32 via 4x2 WMMA 16x16x32 tiles. Staging is a pure
// f16 copy -> async global->LDS when available.
// =====================================================================
template <bool OUT_F32>
__global__ __launch_bounds__(256, 2)
void gemm_wmma_kernel(const _Float16* __restrict__ A,
                      const _Float16* __restrict__ W,
                      const float* __restrict__ bias,
                      void* __restrict__ Cptr) {
    constexpr int BM = 128, BN = 128, BK = 32;
    __shared__ _Float16 As[BM * BK];   // [row][k] row-major
    __shared__ _Float16 Bs[BN * BK];   // [n-row of W][k]

    const int tid  = threadIdx.x;
    const int wave = tid >> 5;
    const int lane = tid & 31;
    const int ln   = lane & 15;
    const int lh   = lane >> 4;
    const int wm   = wave >> 2;        // 0..1  (64 rows each)
    const int wn   = wave & 3;         // 0..3  (32 cols each)

    const int m0 = blockIdx.y * BM;
    const int n0 = blockIdx.x * BN;

    const int srow = tid >> 1;         // staging row 0..127
    const int scg  = (tid & 1) * 16;   // staging col group 0 / 16

    v8f acc[4][2];
#pragma unroll
    for (int mt = 0; mt < 4; ++mt)
#pragma unroll
        for (int nt = 0; nt < 2; ++nt) acc[mt][nt] = (v8f){};

    const _Float16* aRow = A + (size_t)(m0 + srow) * DMODEL + scg;
    const _Float16* wRow = W + (size_t)(n0 + srow) * DMODEL + scg;
    _Float16* aDst = As + srow * BK + scg;
    _Float16* bDst = Bs + srow * BK + scg;

    for (int k0 = 0; k0 < DMODEL; k0 += BK) {
        // ---- stage A/B tiles (pure f16 copy; async path if available) ----
        stage_b128(aRow + k0,     aDst);
        stage_b128(aRow + k0 + 8, aDst + 8);
        stage_b128(wRow + k0,     bDst);
        stage_b128(wRow + k0 + 8, bDst + 8);
        if (k0 + BK < DMODEL) {    // global_prefetch_b8 for next tiles
            __builtin_prefetch(aRow + k0 + BK, 0, 3);
            __builtin_prefetch(wRow + k0 + BK, 0, 3);
        }
        stage_fence();

        // ---- fragments from LDS ----
        // A fragment: lane ln = row, element e -> k = 16*(e>=8) + lh*8 + e%8
        v16h afrag[4];
#pragma unroll
        for (int mt = 0; mt < 4; ++mt) {
            const _Float16* p = As + (wm * 64 + mt * 16 + ln) * BK;
            v8h lo = *(const v8h*)(p + lh * 8);
            v8h hi = *(const v8h*)(p + 16 + lh * 8);
            afrag[mt] = combine16(lo, hi);
        }
        // B fragment: lane ln = column, element e -> k = 16*(e>=8) + lh*8 + e%8
        v16h bfrag[2];
#pragma unroll
        for (int nt = 0; nt < 2; ++nt) {
            const _Float16* p = Bs + (wn * 32 + nt * 16 + ln) * BK + lh * 8;
            bfrag[nt] = combine16(*(const v8h*)p, *(const v8h*)(p + 16));
        }
#pragma unroll
        for (int mt = 0; mt < 4; ++mt)
#pragma unroll
            for (int nt = 0; nt < 2; ++nt)
                acc[mt][nt] = wmma_f16(afrag[mt], bfrag[nt], acc[mt][nt]);
        __syncthreads();
    }

    // ---- epilogue: C[m,n] layout (m = base + 8*lh + i, n = base + ln) ----
#pragma unroll
    for (int mt = 0; mt < 4; ++mt) {
#pragma unroll
        for (int nt = 0; nt < 2; ++nt) {
            const int n  = n0 + wn * 32 + nt * 16 + ln;
            const float bv = bias[n];
#pragma unroll
            for (int i = 0; i < 8; ++i) {
                const int m = m0 + wm * 64 + mt * 16 + lh * 8 + i;
                const float v = acc[mt][nt][i] + bv;
                if (OUT_F32)
                    ((float*)Cptr)[(size_t)m * DMODEL + n] = v;
                else
                    ((_Float16*)Cptr)[(size_t)m * DMODEL + n] = (_Float16)v;
            }
        }
    }
}

// =====================================================================
// RoPE (in-place on f16 [B*S, DMODEL] buffer). One thread per (even,odd)
// pair. Pair index p in [0,1024): d = 2p % 128, freq index jf = p % 64.
// =====================================================================
__global__ __launch_bounds__(256)
void rope_kernel(_Float16* __restrict__ buf) {
    const size_t idx = (size_t)blockIdx.x * blockDim.x + threadIdx.x;  // B*S*1024
    const size_t m = idx >> 10;          // row in [0, 4096)
    const int    p = (int)(idx & 1023);
    const int    s = (int)(m & (SEQ - 1));
    const int    jf = p & 63;
    // inv_freq = 10000^(-2*jf/128) = exp(-jf * ln(10000)/64)
    const float inv = __expf(-0.14391156f * (float)jf);
    const float ang = (float)s * inv;
    float sn, cs;
    __sincosf(ang, &sn, &cs);
    _Float16* ptr = buf + (m << 11) + 2 * p;
    const float e = (float)ptr[0];
    const float o = (float)ptr[1];
    ptr[0] = (_Float16)(e * cs - o * sn);
    ptr[1] = (_Float16)(e * sn + o * cs);
}

// =====================================================================
// Flash attention with causal mask. Grid: (S/128 q-tiles, B*H).
// 256 threads = 8 waves; wave w owns 16 query rows (q0 + 16w + ln).
// Streams key tiles of 64: scores via WMMA (K async-staged in LDS),
// online softmax with shfl_xor half-wave reductions, P staged through
// LDS, PV via WMMA (V staged transposed in LDS). f16 in/out, fp32 accum.
// =====================================================================
__global__ __launch_bounds__(256, 1)
void flash_attn_kernel(const _Float16* __restrict__ Q,
                       const _Float16* __restrict__ K,
                       const _Float16* __restrict__ V,
                       _Float16* __restrict__ O) {
    __shared__ _Float16 Kt[64 * DK];       // [key][d]       16 KB
    __shared__ _Float16 Vt[DK * 64];       // [d][key]       16 KB (transposed)
    __shared__ _Float16 Pst[8 * 16 * 64];  // per-wave P     16 KB

    const int tid  = threadIdx.x;
    const int wave = tid >> 5;
    const int lane = tid & 31;
    const int ln   = lane & 15;
    const int lh   = lane >> 4;

    const int q0 = blockIdx.x * 128;
    const int bh = blockIdx.y;
    const int b  = bh >> 4;
    const int h  = bh & 15;

    // ---- load Q fragments for this wave's 16 rows (held all kernel) ----
    const size_t qbase = (size_t)(b * SEQ + q0 + wave * 16 + ln) * DMODEL + h * DK;
    v16h qf[4];
#pragma unroll
    for (int dk = 0; dk < 4; ++dk) {
        v8h lo = *(const v8h*)(Q + qbase + dk * 32 + lh * 8);
        v8h hi = *(const v8h*)(Q + qbase + dk * 32 + 16 + lh * 8);
        qf[dk] = combine16(lo, hi);
    }

    v8f oacc[8];
#pragma unroll
    for (int dt = 0; dt < 8; ++dt) oacc[dt] = (v8f){};
    float mst[8], lst[8];
#pragma unroll
    for (int i = 0; i < 8; ++i) { mst[i] = -1e30f; lst[i] = 0.0f; }

    const float scale = 0.0883883476f;   // 1/sqrt(128)
    const int rbase = q0 + wave * 16 + lh * 8;
    const int kend  = q0 + 128;          // causal bound

    for (int k0 = 0; k0 < kend; k0 += 64) {
        __syncthreads();   // protect Kt/Vt against readers of prev iter
        // ---- stage K tile [64 keys][128 d] (async copy path) ----
        {
            const int row = tid >> 2;
            const int ds  = (tid & 3) * 32;
            const _Float16* src = K + (size_t)(b * SEQ + k0 + row) * DMODEL + h * DK + ds;
            _Float16* dst = Kt + row * DK + ds;
#pragma unroll
            for (int j = 0; j < 4; ++j) stage_b128(src + 8 * j, dst + 8 * j);
            if (k0 + 64 < kend)
                __builtin_prefetch(src + 64 * DMODEL, 0, 3);
        }
        // ---- stage V tile transposed: Vt[d][key] ----
        {
            const int kk = tid >> 2;
            const int ds = (tid & 3) * 32;
            const v8h* src = (const v8h*)(V + (size_t)(b * SEQ + k0 + kk) * DMODEL + h * DK + ds);
#pragma unroll
            for (int j = 0; j < 4; ++j) {
                v8h vv = src[j];
#pragma unroll
                for (int e = 0; e < 8; ++e) Vt[(ds + 8 * j + e) * 64 + kk] = vv[e];
            }
            if (k0 + 64 < kend)
                __builtin_prefetch((const _Float16*)src + 64 * DMODEL, 0, 3);
        }
        stage_fence();

        // ---- scores S = Q K^T : 16 rows x 64 cols per wave ----
        v8f sa[4];
#pragma unroll
        for (int nt = 0; nt < 4; ++nt) sa[nt] = (v8f){};
#pragma unroll
        for (int dk = 0; dk < 4; ++dk) {
#pragma unroll
            for (int nt = 0; nt < 4; ++nt) {
                // B fragment: column = key (nt*16+ln), element e -> d = dk*32 + 16*(e>=8) + lh*8 + e%8
                const _Float16* p = Kt + (nt * 16 + ln) * DK + dk * 32 + lh * 8;
                v16h kb = combine16(*(const v8h*)p, *(const v8h*)(p + 16));
                sa[nt] = wmma_f16(qf[dk], kb, sa[nt]);
            }
        }

        // ---- scale + causal mask + row max ----
        float rmax[8];
#pragma unroll
        for (int i = 0; i < 8; ++i) rmax[i] = -1e30f;
#pragma unroll
        for (int nt = 0; nt < 4; ++nt) {
            const int c = k0 + nt * 16 + ln;
#pragma unroll
            for (int i = 0; i < 8; ++i) {
                float s = sa[nt][i] * scale;
                if (c > rbase + i) s = -1e30f;
                sa[nt][i] = s;
                rmax[i] = fmaxf(rmax[i], s);
            }
        }
        // reduce across the 16 lanes of each half (rows differ per half)
#pragma unroll
        for (int off = 1; off < 16; off <<= 1)
#pragma unroll
            for (int i = 0; i < 8; ++i)
                rmax[i] = fmaxf(rmax[i], __shfl_xor(rmax[i], off, 32));

        // ---- online softmax update ----
        float escale[8], rsum[8];
#pragma unroll
        for (int i = 0; i < 8; ++i) {
            const float mn = fmaxf(mst[i], rmax[i]);
            escale[i] = __expf(mst[i] - mn);
            mst[i] = mn;
            rsum[i] = 0.0f;
        }
#pragma unroll
        for (int nt = 0; nt < 4; ++nt)
#pragma unroll
            for (int i = 0; i < 8; ++i) {
                const float pe = __expf(sa[nt][i] - mst[i]);
                sa[nt][i] = pe;
                rsum[i] += pe;
            }
#pragma unroll
        for (int off = 1; off < 16; off <<= 1)
#pragma unroll
            for (int i = 0; i < 8; ++i)
                rsum[i] += __shfl_xor(rsum[i], off, 32);
#pragma unroll
        for (int i = 0; i < 8; ++i) lst[i] = lst[i] * escale[i] + rsum[i];
#pragma unroll
        for (int dt = 0; dt < 8; ++dt)
#pragma unroll
            for (int i = 0; i < 8; ++i) oacc[dt][i] *= escale[i];

        // ---- stage P (f16) into this wave's private LDS region ----
        _Float16* Pw = Pst + wave * (16 * 64);
#pragma unroll
        for (int nt = 0; nt < 4; ++nt)
#pragma unroll
            for (int i = 0; i < 8; ++i)
                Pw[(lh * 8 + i) * 64 + nt * 16 + ln] = (_Float16)sa[nt][i];
        // same-wave LDS ops are in-order: loads below see the stores above

        // ---- O += P V  (A = P rows, B = V columns from Vt) ----
#pragma unroll
        for (int ks = 0; ks < 2; ++ks) {
            const _Float16* pp = Pw + ln * 64 + ks * 32;
            v8h plo = *(const v8h*)(pp + lh * 8);
            v8h phi = *(const v8h*)(pp + 16 + lh * 8);
            v16h pf = combine16(plo, phi);
#pragma unroll
            for (int dt = 0; dt < 8; ++dt) {
                const _Float16* vp = Vt + (dt * 16 + ln) * 64 + ks * 32 + lh * 8;
                v16h vf = combine16(*(const v8h*)vp, *(const v8h*)(vp + 16));
                oacc[dt] = wmma_f16(pf, vf, oacc[dt]);
            }
        }
    }

    // ---- epilogue: out = O / l, write f16 [B*S, DMODEL] ----
#pragma unroll
    for (int i = 0; i < 8; ++i) lst[i] = 1.0f / lst[i];
#pragma unroll
    for (int dt = 0; dt < 8; ++dt) {
        const int d = dt * 16 + ln;
#pragma unroll
        for (int i = 0; i < 8; ++i) {
            const int r = rbase + i;
            O[(size_t)(b * SEQ + r) * DMODEL + h * DK + d] = (_Float16)(oacc[dt][i] * lst[i]);
        }
    }
}

// =====================================================================
// host-side launch
// =====================================================================
extern "C" void kernel_launch(void* const* d_in, const int* in_sizes, int n_in,
                              void* d_out, int out_size, void* d_ws, size_t ws_size,
                              hipStream_t stream) {
    (void)in_sizes; (void)n_in; (void)out_size; (void)ws_size;
    const float* x   = (const float*)d_in[0];
    const float* wqw = (const float*)d_in[1];
    const float* wqb = (const float*)d_in[2];
    const float* wkw = (const float*)d_in[3];
    const float* wkb = (const float*)d_in[4];
    const float* wvw = (const float*)d_in[5];
    const float* wvb = (const float*)d_in[6];
    const float* wow = (const float*)d_in[7];
    const float* wob = (const float*)d_in[8];

    char* ws = (char*)d_ws;
    const size_t XSZ = (size_t)MROWS  * DMODEL * sizeof(_Float16);   // 16 MB
    const size_t WSZ = (size_t)DMODEL * DMODEL * sizeof(_Float16);   //  8 MB
    _Float16* xh  = (_Float16*)(ws);
    _Float16* wqh = (_Float16*)(ws + XSZ);
    _Float16* wkh = (_Float16*)(ws + XSZ + WSZ);
    _Float16* wvh = (_Float16*)(ws + XSZ + 2 * WSZ);
    _Float16* woh = (_Float16*)(ws + XSZ + 3 * WSZ);
    _Float16* Qb  = (_Float16*)(ws + XSZ + 4 * WSZ);
    _Float16* Kb  = (_Float16*)(ws + 2 * XSZ + 4 * WSZ);
    _Float16* Vb  = (_Float16*)(ws + 3 * XSZ + 4 * WSZ);
    _Float16* Ab  = (_Float16*)(ws + 4 * XSZ + 4 * WSZ);

    // one-shot fp32 -> f16 conversion (x + 4 weight matrices)
    const int xblocks = (int)((size_t)MROWS * DMODEL / 8 / 256);     // 4096
    const int wblocks = (int)((size_t)DMODEL * DMODEL / 8 / 256);    // 2048
    f32_to_f16_kernel<<<xblocks, 256, 0, stream>>>(x,   xh);
    f32_to_f16_kernel<<<wblocks, 256, 0, stream>>>(wqw, wqh);
    f32_to_f16_kernel<<<wblocks, 256, 0, stream>>>(wkw, wkh);
    f32_to_f16_kernel<<<wblocks, 256, 0, stream>>>(wvw, wvh);
    f32_to_f16_kernel<<<wblocks, 256, 0, stream>>>(wow, woh);

    dim3 ggrid(DMODEL / 128, MROWS / 128);   // (16, 32)
    dim3 gblk(256);
    gemm_wmma_kernel<false><<<ggrid, gblk, 0, stream>>>(xh, wqh, wqb, Qb);
    gemm_wmma_kernel<false><<<ggrid, gblk, 0, stream>>>(xh, wkh, wkb, Kb);
    gemm_wmma_kernel<false><<<ggrid, gblk, 0, stream>>>(xh, wvh, wvb, Vb);

    const int npairs_blocks = (MROWS * (DMODEL / 2)) / 256;          // 16384
    rope_kernel<<<npairs_blocks, 256, 0, stream>>>(Qb);
    rope_kernel<<<npairs_blocks, 256, 0, stream>>>(Kb);

    dim3 agrid(SEQ / 128, BATCH * NHEADS);   // (16, 32)
    flash_attn_kernel<<<agrid, 256, 0, stream>>>(Qb, Kb, Vb, Ab);

    gemm_wmma_kernel<true><<<ggrid, gblk, 0, stream>>>(Ab, woh, wob, (float*)d_out);
}
